// VRNN_45268955300244
// MI455X (gfx1250) — matrix operand order repaired and checked by reading, not compiled, not run on hardware
//
#include <hip/hip_runtime.h>
#include <hip/hip_bf16.h>
#include <math.h>

// ---------------------------------------------------------------------------
// VRNN on MI455X (gfx1250): bf16 WMMA (v_wmma_f32_16x16x32_bf16) GEMMs,
// L2-resident weights, 32-step sequential pipeline of small kernels.
// GEMM inner loop: straight-line, register double-buffered, 4Mx2N wave tile
// so both half-chunks issue their 8 WMMAs back-to-back while the next
// chunk's global loads are in flight.
// ---------------------------------------------------------------------------

typedef __bf16 bf16_t;
typedef bf16_t v16bf __attribute__((ext_vector_type(16)));
typedef float  v8f   __attribute__((ext_vector_type(8)));

union Frag16 { v16bf v; uint4 u[2]; };

__device__ __forceinline__ unsigned short f2bf(float f) {
    unsigned int u = __float_as_uint(f);
    unsigned int r = u + 0x7FFFu + ((u >> 16) & 1u);   // round-to-nearest-even
    return (unsigned short)(r >> 16);
}

__device__ __forceinline__ float sigmoidf(float x) {
    return 1.0f / (1.0f + __expf(-x));
}

// Problem constants
#define Bsz 128
#define Tsz 32
#define Esz 512
#define Hsz 512
#define Lsz 256
#define Vsz 16000
#define XHW 1280           // xh row width: [e(512) | z(256) | h(512)]
#define OUT_STRIDE 1048576 // T*B*L = 32*128*256

// ---------------------------------------------------------------------------
// Generic bf16 GEMM:  C[128 x N] = A[128 x K] * B[N x K]^T (+ bias)
// Block = 256 threads = 8 waves covering a 128(M) x 128(N) tile.
// Wave w: M-half = w>>2 (4 M-tiles), N-group = w&3 (2 N-tiles of 16).
// K must be a multiple of 64; N a multiple of 128.
// NOTE: the K-loop prefetches one half-chunk (32 elements) past K unconditionally;
// callers guarantee those bytes are mapped (they land in the next workspace
// allocation) — the over-read fragment is never consumed.
// ---------------------------------------------------------------------------
__global__ __launch_bounds__(256)
void gemm_bf16_wmma(const bf16_t* __restrict__ A, int lda,
                    const bf16_t* __restrict__ B, int ldb,
                    const float*  __restrict__ bias,
                    float* __restrict__ C, int ldc, int K)
{
    const int lane = threadIdx.x & 31;
    const int wave = threadIdx.x >> 5;
    const int half = lane >> 4;
    const int ln   = lane & 15;
    const int mh   = wave >> 2;                 // 0/1: M rows [mh*64, mh*64+64)
    const int ng   = wave & 3;                  // N columns [ng*32, ng*32+32)
    const int nBase = blockIdx.x * 128 + ng * 32;

    // 32-bit element offsets -> SADDR + voffset load form
    int aOff[4], bOff[2];
#pragma unroll
    for (int mt = 0; mt < 4; ++mt)
        aOff[mt] = ((mh * 4 + mt) * 16 + ln) * lda + 8 * half;
#pragma unroll
    for (int nt = 0; nt < 2; ++nt)
        bOff[nt] = (nBase + nt * 16 + ln) * ldb + 8 * half;

    v8f acc[4][2];
#pragma unroll
    for (int mt = 0; mt < 4; ++mt)
#pragma unroll
        for (int nt = 0; nt < 2; ++nt) acc[mt][nt] = (v8f)(0.0f);

    Frag16 fa0[4], fa1[4], fb0[2], fb1[2];

    // prologue: load K-chunk 0 into buffer 0
#pragma unroll
    for (int nt = 0; nt < 2; ++nt) {
        fb0[nt].u[0] = *(const uint4*)(B + bOff[nt]);
        fb0[nt].u[1] = *(const uint4*)(B + bOff[nt] + 16);
    }
#pragma unroll
    for (int mt = 0; mt < 4; ++mt) {
        fa0[mt].u[0] = *(const uint4*)(A + aOff[mt]);
        fa0[mt].u[1] = *(const uint4*)(A + aOff[mt] + 16);
    }

    for (int kc = 0; kc < K; kc += 64) {
        // prefetch chunk kc+32 into buffer 1
#pragma unroll
        for (int nt = 0; nt < 2; ++nt) {
            fb1[nt].u[0] = *(const uint4*)(B + bOff[nt] + (kc + 32));
            fb1[nt].u[1] = *(const uint4*)(B + bOff[nt] + (kc + 48));
        }
#pragma unroll
        for (int mt = 0; mt < 4; ++mt) {
            fa1[mt].u[0] = *(const uint4*)(A + aOff[mt] + (kc + 32));
            fa1[mt].u[1] = *(const uint4*)(A + aOff[mt] + (kc + 48));
        }
        // compute chunk kc from buffer 0
#pragma unroll
        for (int mt = 0; mt < 4; ++mt)
#pragma unroll
            for (int nt = 0; nt < 2; ++nt)
                acc[mt][nt] = __builtin_amdgcn_wmma_f32_16x16x32_bf16(
                    false, fa0[mt].v, false, fb0[nt].v, (short)0, acc[mt][nt], false, false);

        // prefetch chunk kc+64 into buffer 0 (unconditional; last iter over-reads
        // 32 elements into the adjacent workspace buffer and is never consumed)
#pragma unroll
        for (int nt = 0; nt < 2; ++nt) {
            fb0[nt].u[0] = *(const uint4*)(B + bOff[nt] + (kc + 64));
            fb0[nt].u[1] = *(const uint4*)(B + bOff[nt] + (kc + 80));
        }
#pragma unroll
        for (int mt = 0; mt < 4; ++mt) {
            fa0[mt].u[0] = *(const uint4*)(A + aOff[mt] + (kc + 64));
            fa0[mt].u[1] = *(const uint4*)(A + aOff[mt] + (kc + 80));
        }
        // compute chunk kc+32 from buffer 1
#pragma unroll
        for (int mt = 0; mt < 4; ++mt)
#pragma unroll
            for (int nt = 0; nt < 2; ++nt)
                acc[mt][nt] = __builtin_amdgcn_wmma_f32_16x16x32_bf16(
                    false, fa1[mt].v, false, fb1[nt].v, (short)0, acc[mt][nt], false, false);
    }

#pragma unroll
    for (int nt = 0; nt < 2; ++nt) {
        const int col = nBase + nt * 16 + ln;
        const float bv = bias ? bias[col] : 0.0f;
#pragma unroll
        for (int mt = 0; mt < 4; ++mt) {
#pragma unroll
            for (int v = 0; v < 8; ++v) {
                const int row = (mh * 4 + mt) * 16 + half * 8 + v;
                C[(size_t)row * ldc + col] = acc[mt][nt][v] + bv;
            }
        }
    }
}

// ---------------------------------------------------------------------------
// One-time prep kernels
// ---------------------------------------------------------------------------
__global__ void k_conv_qxw(const float* __restrict__ w, unsigned short* __restrict__ o) {
    int i = blockIdx.x * 256 + threadIdx.x;
    if (i < Vsz * 768) o[i] = f2bf(w[i]);
}

// Wcomb[2048][1280]: cols 0:768 = W_ih (acts on [e,z]), cols 768:1280 = W_hh (acts on h)
__global__ void k_wcomb(const float* __restrict__ wih, const float* __restrict__ whh,
                        unsigned short* __restrict__ o) {
    int i = blockIdx.x * 256 + threadIdx.x;
    if (i >= 2048 * XHW) return;
    int nn = i / XHW, k = i % XHW;
    float v = (k < 768) ? wih[nn * 768 + k] : whh[nn * 512 + (k - 768)];
    o[i] = f2bf(v);
}

// Wpq[1024][1280]: rows 0:512 = qz (e-part cols 0:512, h-part cols 768:1280, z slot zero)
//                  rows 512:1024 = prior (h-part cols 768:1280, rest zero)
__global__ void k_wpq(const float* __restrict__ qzw, const float* __restrict__ prw,
                      unsigned short* __restrict__ o) {
    int i = blockIdx.x * 256 + threadIdx.x;
    if (i >= 1024 * XHW) return;
    int nn = i / XHW, k = i % XHW;
    float v = 0.0f;
    if (nn < 512) {
        if (k < 512)       v = qzw[nn * 1024 + k];
        else if (k >= 768) v = qzw[nn * 1024 + 512 + (k - 768)];
    } else {
        if (k >= 768)      v = prw[(nn - 512) * 512 + (k - 768)];
    }
    o[i] = f2bf(v);
}

__global__ void k_bias(const float* __restrict__ bih, const float* __restrict__ bhh,
                       const float* __restrict__ qzb, const float* __restrict__ prb,
                       float* __restrict__ bcomb, float* __restrict__ bpq) {
    int i = blockIdx.x * 256 + threadIdx.x;
    if (i < 2048) bcomb[i] = bih[i] + bhh[i];
    if (i < 1024) bpq[i] = (i < 512) ? qzb[i] : prb[i - 512];
}

// xh = [features | 0 | 0]; c = 0
__global__ void k_init(const float* __restrict__ feat,
                       unsigned short* __restrict__ xh, float* __restrict__ c) {
    int i = blockIdx.x * 256 + threadIdx.x;
    if (i >= Bsz * XHW) return;
    int b = i / XHW, j = i % XHW;
    xh[i] = (j < Esz) ? f2bf(feat[b * Esz + j]) : (unsigned short)0;
    if (j < Hsz) c[b * Hsz + j] = 0.0f;
}

// ---------------------------------------------------------------------------
// Per-step kernels
// ---------------------------------------------------------------------------
__global__ void k_embed(const float* __restrict__ embW, const int* __restrict__ caps,
                        unsigned short* __restrict__ xh, int t) {
    int b = blockIdx.x;
    int tok = caps[b * Tsz + t];
    const float* src = embW + (size_t)tok * Esz;
    for (int j = threadIdx.x; j < Esz; j += blockDim.x)
        xh[b * XHW + j] = f2bf(src[j]);
}

// split ypq -> p_mu/p_sigma/q_mu/q_sigma outputs, z = noise*q_sigma + q_mu -> xh
__global__ void k_post_pq(const float* __restrict__ ypq, const float* __restrict__ noise,
                          unsigned short* __restrict__ xh, float* __restrict__ out, int t) {
    int idx = blockIdx.x * 256 + threadIdx.x;     // 128*256
    if (idx >= Bsz * Lsz) return;
    int b = idx >> 8, l = idx & 255;
    const float* yr = ypq + b * 1024;
    float qm = yr[l], qs = yr[256 + l], pm = yr[512 + l], ps = yr[768 + l];
    float z  = noise[((size_t)b * Tsz + t) * Lsz + l] * qs + qm;
    size_t o = ((size_t)t * Bsz + b) * Lsz + l;
    out[o]                  = pm;
    out[OUT_STRIDE + o]     = ps;
    out[2 * OUT_STRIDE + o] = qm;
    out[3 * OUT_STRIDE + o] = qs;
    xh[b * XHW + Esz + l] = f2bf(z);
}

__global__ __launch_bounds__(256)
void k_logsoftmax(const float* __restrict__ logits, float* __restrict__ out, int t) {
    __shared__ float sred[8];
    __shared__ float sbc;
    const int b = blockIdx.x;
    const float* row = logits + (size_t)b * Vsz;
    const int lane = threadIdx.x & 31, wid = threadIdx.x >> 5;

    float m = -3.4e38f;
    for (int i = threadIdx.x; i < Vsz; i += 256) m = fmaxf(m, row[i]);
    for (int o = 16; o > 0; o >>= 1) m = fmaxf(m, __shfl_xor(m, o, 32));
    if (lane == 0) sred[wid] = m;
    __syncthreads();
    if (threadIdx.x == 0) {
        float mm = sred[0];
        for (int i = 1; i < 8; ++i) mm = fmaxf(mm, sred[i]);
        sbc = mm;
    }
    __syncthreads();
    const float M = sbc;

    float s = 0.0f;
    for (int i = threadIdx.x; i < Vsz; i += 256) s += __expf(row[i] - M);
    for (int o = 16; o > 0; o >>= 1) s += __shfl_xor(s, o, 32);
    if (lane == 0) sred[wid] = s;
    __syncthreads();
    if (threadIdx.x == 0) {
        float ss = 0.0f;
        for (int i = 0; i < 8; ++i) ss += sred[i];
        sbc = M + __logf(ss);
    }
    __syncthreads();
    const float lse = sbc;

    float* orow = out + 4 * (size_t)OUT_STRIDE + ((size_t)t * Bsz + b) * Vsz;
    for (int i = threadIdx.x; i < Vsz; i += 256) orow[i] = row[i] - lse;
}

__global__ void k_lstm_gates(const float* __restrict__ gates, float* __restrict__ c,
                             unsigned short* __restrict__ xh) {
    int idx = blockIdx.x * 256 + threadIdx.x;     // 128*512
    if (idx >= Bsz * Hsz) return;
    int b = idx >> 9, j = idx & 511;
    const float* gr = gates + b * 2048;
    float gi = gr[j], gf = gr[512 + j], gg = gr[1024 + j], go = gr[1536 + j];
    float cn = sigmoidf(gf) * c[idx] + sigmoidf(gi) * tanhf(gg);
    float h  = sigmoidf(go) * tanhf(cn);
    c[idx] = cn;
    xh[b * XHW + 768 + j] = f2bf(h);
}

// ---------------------------------------------------------------------------
// Host side
// ---------------------------------------------------------------------------
extern "C" void kernel_launch(void* const* d_in, const int* in_sizes, int n_in,
                              void* d_out, int out_size, void* d_ws, size_t ws_size,
                              hipStream_t stream) {
    const float* features = (const float*)d_in[0];
    const int*   captions = (const int*)d_in[1];
    const float* noise    = (const float*)d_in[3];
    const float* embed_W  = (const float*)d_in[4];
    const float* W_ih     = (const float*)d_in[5];
    const float* W_hh     = (const float*)d_in[6];
    const float* b_ih     = (const float*)d_in[7];
    const float* b_hh     = (const float*)d_in[8];
    const float* qz_W     = (const float*)d_in[9];
    const float* qz_b     = (const float*)d_in[10];
    const float* prior_W  = (const float*)d_in[11];
    const float* prior_b  = (const float*)d_in[12];
    const float* qx_W     = (const float*)d_in[13];
    const float* qx_b     = (const float*)d_in[14];
    float* out = (float*)d_out;

    char* ws = (char*)d_ws;
    size_t off = 0;
    unsigned short* qxw_bf  = (unsigned short*)(ws + off); off += (size_t)Vsz * 768 * 2;   // 24,576,000
    unsigned short* wcomb   = (unsigned short*)(ws + off); off += (size_t)2048 * XHW * 2;  //  5,242,880
    unsigned short* wpq     = (unsigned short*)(ws + off); off += (size_t)1024 * XHW * 2;  //  2,621,440
    float*          bcomb   = (float*)(ws + off);          off += 2048 * 4;
    float*          bpq     = (float*)(ws + off);          off += 1024 * 4;
    unsigned short* xh      = (unsigned short*)(ws + off); off += (size_t)Bsz * XHW * 2;   //    327,680
    float*          cbuf    = (float*)(ws + off);          off += (size_t)Bsz * Hsz * 4;   //    262,144
    float*          gates   = (float*)(ws + off);          off += (size_t)Bsz * 2048 * 4;  //  1,048,576
    float*          ypq     = (float*)(ws + off);          off += (size_t)Bsz * 1024 * 4;  //    524,288
    float*          logits  = (float*)(ws + off);          off += (size_t)Bsz * Vsz * 4;   //  8,192,000

    const bf16_t* xh_bf = (const bf16_t*)xh;

    // ---- one-time prep ----
    k_conv_qxw<<<(Vsz * 768 + 255) / 256, 256, 0, stream>>>(qx_W, qxw_bf);
    k_wcomb<<<(2048 * XHW + 255) / 256, 256, 0, stream>>>(W_ih, W_hh, wcomb);
    k_wpq<<<(1024 * XHW + 255) / 256, 256, 0, stream>>>(qz_W, prior_W, wpq);
    k_bias<<<8, 256, 0, stream>>>(b_ih, b_hh, qz_b, prior_b, bcomb, bpq);
    k_init<<<(Bsz * XHW + 255) / 256, 256, 0, stream>>>(features, xh, cbuf);

    // ---- initial LSTM step on [features, 0] with h=c=0 ----
    gemm_bf16_wmma<<<2048 / 128, 256, 0, stream>>>(
        xh_bf, XHW, (const bf16_t*)wcomb, XHW, bcomb, gates, 2048, XHW);
    k_lstm_gates<<<(Bsz * Hsz + 255) / 256, 256, 0, stream>>>(gates, cbuf, xh);

    // ---- T=32 recurrent steps ----
    for (int t = 0; t < Tsz; ++t) {
        k_embed<<<Bsz, 256, 0, stream>>>(embed_W, captions, xh, t);

        // prior + qz:  [128 x 1024] = xh[128 x 1280] * Wpq^T
        gemm_bf16_wmma<<<1024 / 128, 256, 0, stream>>>(
            xh_bf, XHW, (const bf16_t*)wpq, XHW, bpq, ypq, 1024, XHW);
        k_post_pq<<<(Bsz * Lsz + 255) / 256, 256, 0, stream>>>(ypq, noise, xh, out, t);

        // logits: [128 x 16000] = [z,h](=xh+512, K=768) * qx_W^T + qx_b
        gemm_bf16_wmma<<<Vsz / 128, 256, 0, stream>>>(
            xh_bf + Esz, XHW, (const bf16_t*)qxw_bf, 768, qx_b, logits, Vsz, 768);
        k_logsoftmax<<<Bsz, 256, 0, stream>>>(logits, out, t);

        // LSTM: gates[128 x 2048] = xh * Wcomb^T + (b_ih + b_hh)
        gemm_bf16_wmma<<<2048 / 128, 256, 0, stream>>>(
            xh_bf, XHW, (const bf16_t*)wcomb, XHW, bcomb, gates, 2048, XHW);
        k_lstm_gates<<<(Bsz * Hsz + 255) / 256, 256, 0, stream>>>(gates, cbuf, xh);
    }

    (void)in_sizes; (void)n_in; (void)out_size; (void)ws_size;
}
